// ConvTransformerEncoder_52012053955076
// MI455X (gfx1250) — matrix operand order repaired and checked by reading, not compiled, not run on hardware
//
#include <hip/hip_runtime.h>

// ---------------------------------------------------------------------------
// ConvTransformerEncoder on gfx1250 (MI455X).
//
// Key algebraic fact exploited (exact): the attention conv is linear over the
// concatenated [q;k] channels, so logits(lq,lk) = Aq(q[lq]) + Ak(k[lk]) + b.
// Softmax is taken over lk; the Aq(q)+b term is constant in lk and cancels.
// => q_w/q_b and the q-half of att_w are mathematically dead, and the head
// output is independent of lq (broadcast over frames).
//
// Heavy convs (kv 32->32, ff 128->128, 3x3 SAME) run as implicit GEMM on
// v_wmma_f32_16x16x32_f16: one wave owns a 16(cout)x16(pixel) tile, K=Cin*9
// consumed 32 at a time (cin-block x single 3x3 tap). LDS activation layout
// [row][col][ch] makes each B fragment a contiguous 32B LDS read in this
// code's contiguous k order (k = 16h + s, same as A, not the ISA table order); weights are device-prepacked into A-fragment order so
// each A fragment is a contiguous 32B global load.
// ---------------------------------------------------------------------------

typedef __attribute__((ext_vector_type(16))) _Float16 v16h;
typedef __attribute__((ext_vector_type(8)))  float    v8f;

#define NLAYERS 5
#define NHEADS  4
#define DHC     32
#define CCH     128
#define BB      2
#define TT      12
#define HWDIM   32
#define NPIX    1024                 // 32*32
#define NIMG    (BB*TT)              // 24
#define IMG_STRIDE ((long)CCH*NPIX)  // 131072
#define TOT_ELEMS (NIMG*CCH*NPIX)    // 3145728

// ---------------------------------------------------------------------------
// Weight pre-packing into WMMA A-fragment lane layout.
// A (16x32 f16), this code's k order (not ISA table): lane<16 holds row M=lane, halves j: K = j;
//                lane>=16 holds row M=lane-16,      K = j+16.
// Fragment storage: [... mtile, cinBlock, tap ...][lane][16 halves] contiguous.
// ---------------------------------------------------------------------------
__device__ __forceinline__ int a_kmap(int lane, int j) {
    return (lane < 16) ? j
                       : j + 16;
}

// kv weights: (L,H,32,32,3,3) -> wp[((lh*2+mt)*9+t)*512 + lane*16 + j]
__global__ void pack_head_w(const float* __restrict__ kvw, _Float16* __restrict__ wp) {
    int idx = blockIdx.x * blockDim.x + threadIdx.x;
    const int total = NLAYERS * NHEADS * 2 * 9 * 32 * 16;
    if (idx >= total) return;
    int j    = idx & 15;
    int lane = (idx >> 4) & 31;
    int rest = idx >> 9;
    int t  = rest % 9;  rest /= 9;
    int mt = rest & 1;  rest >>= 1;
    int lh = rest;                       // li*NHEADS + head
    int m   = mt * 16 + (lane & 15);
    int cin = a_kmap(lane, j);
    float w = kvw[(((long)lh * 32 + m) * 32 + cin) * 9 + t];
    wp[idx] = (_Float16)w;
}

// ff weights: (L,128,128,3,3) -> wp[(((li*8+mt)*4+cb)*9+t)*512 + lane*16 + j]
__global__ void pack_ff_w(const float* __restrict__ fw, _Float16* __restrict__ wp) {
    int idx = blockIdx.x * blockDim.x + threadIdx.x;
    const int total = NLAYERS * 8 * 4 * 9 * 32 * 16;
    if (idx >= total) return;
    int j    = idx & 15;
    int lane = (idx >> 4) & 31;
    int rest = idx >> 9;
    int t  = rest % 9;  rest /= 9;
    int cb = rest & 3;  rest >>= 2;
    int mt = rest & 7;
    int li = rest >> 3;
    int m   = mt * 16 + (lane & 15);
    int cin = cb * 32 + a_kmap(lane, j);
    float w = fw[(((long)li * CCH + m) * CCH + cin) * 9 + t];
    wp[idx] = (_Float16)w;
}

__global__ void f32_to_f16(const float* __restrict__ src, _Float16* __restrict__ dst, int n) {
    int i = blockIdx.x * blockDim.x + threadIdx.x;
    if (i < n) dst[i] = (_Float16)src[i];
}

// ---------------------------------------------------------------------------
// Implicit-GEMM 3x3 SAME conv, Cin = 32*cinBlocks, Cout = gridDim.y*NW*16.
// Block = NW waves; each wave owns one 16-row (cout) tile; all waves share the
// LDS-staged activation block [3 rows][18 cols][32 ch] per cin-block.
// grid = (nImages, Cout/16/NW, 64 pixel tiles (32 rows x 2 halves)).
// ---------------------------------------------------------------------------
__global__ void conv3x3_wmma(const _Float16* __restrict__ act, long actImgStride,
                             const v16h* __restrict__ wp,     // packed A fragments
                             const float* __restrict__ bias,
                             float* __restrict__ out32, _Float16* __restrict__ out16,
                             const float* __restrict__ resid, long outImgStride,
                             int cinBlocks, int relu) {
    __shared__ __align__(32) _Float16 sAct[3 * 18 * 32];

    const int img  = blockIdx.x;
    const int nw   = blockDim.x >> 5;
    const int wave = threadIdx.x >> 5;
    const int lane = threadIdx.x & 31;
    const int mtile = blockIdx.y * nw + wave;
    const int y  = blockIdx.z >> 1;
    const int x0 = (blockIdx.z & 1) << 4;

    const _Float16* actImg = act + (long)img * actImgStride;
    v8f acc = {};

    for (int cb = 0; cb < cinBlocks; ++cb) {
        __syncthreads();
        // Cooperative stage: [row 0..2][col 0..17][ch 0..31], zero-padded.
        for (int i = threadIdx.x; i < 3 * 18 * 32; i += blockDim.x) {
            int row = i / (18 * 32);
            int rem = i - row * (18 * 32);
            int col = rem >> 5;
            int ch  = rem & 31;
            int gy = y + row - 1;
            int gx = x0 + col - 1;
            _Float16 v = (_Float16)0.f;
            if (gy >= 0 && gy < HWDIM && gx >= 0 && gx < HWDIM)
                v = actImg[(long)(cb * 32 + ch) * NPIX + gy * HWDIM + gx];
            sAct[i] = v;
        }
        __syncthreads();

        const v16h* wpc = wp + ((long)(mtile * cinBlocks + cb) * 9) * 32;
        const int n = lane & 15;
        const int kb = (lane >> 4) << 4;   // 0 or 16: dense-B K split
#pragma unroll
        for (int t = 0; t < 9; ++t) {
            v16h a = wpc[t * 32 + lane];
            // B fragment: 16 consecutive channels at (row = t/3, col = n + t%3)
            const v16h* bp =
                (const v16h*)&sAct[((t / 3) * 18 + n + (t % 3)) * 32 + kb];
            v16h b = *bp;
            acc = __builtin_amdgcn_wmma_f32_16x16x32_f16(
                false, a, false, b, (short)0, acc, false, false);
        }
    }

    // C/D layout: lane col = lane&15; VGPR r -> row r + 8*(lane>>4).
    const int n = lane & 15;
    const long obase = (long)img * outImgStride;
#pragma unroll
    for (int r = 0; r < 8; ++r) {
        int m = mtile * 16 + r + ((lane >> 4) << 3);
        float v = acc[r] + bias[m];
        if (relu) v = v > 0.f ? v : 0.f;
        long oi = obase + (long)m * NPIX + y * HWDIM + x0 + n;
        if (resid) v += resid[oi];
        if (out32) out32[oi] = v;
        if (out16) out16[oi] = (_Float16)v;
    }
}

// ---------------------------------------------------------------------------
// Attention logits: s[b,hd,t,pix] = ab + sum_{cin,tap} awK * (v + pos(t,cin)),
// zero-padded 3x3 conv of k = v + pos with the K-half of att_w. (q-half cancels
// in the lk-softmax.)  One block per (b,hd,t).
// ---------------------------------------------------------------------------
__global__ void attn_logits(const float* __restrict__ V,
                            const float* __restrict__ att_w,
                            const float* __restrict__ att_b,
                            float* __restrict__ S, int li) {
    __shared__ float w[288];
    __shared__ float posv[32];
    int t  = blockIdx.x % TT;
    int hd = (blockIdx.x / TT) % NHEADS;
    int b  = blockIdx.x / (TT * NHEADS);

    for (int i = threadIdx.x; i < 288; i += blockDim.x) {
        int cin = i / 9, tl = i % 9;
        w[i] = att_w[((long)(li * NHEADS + hd) * 64 + 32 + cin) * 9 + tl];
    }
    if (threadIdx.x < 32) {
        int c = threadIdx.x;
        // pe: even ch -> sin(t*div), odd -> cos(t*div), div = exp(-(c&~1)*ln(1e4)/32)
        float div = __expf(-(float)(c & ~1) * (9.210340371976184f / 32.f));
        posv[c] = (c & 1) ? __cosf((float)t * div) : __sinf((float)t * div);
    }
    __syncthreads();

    float ab = att_b[li * NHEADS + hd];
    const float* vimg = V + ((long)(b * TT + t) * CCH + hd * DHC) * NPIX;
    for (int p = threadIdx.x; p < NPIX; p += blockDim.x) {
        int y = p >> 5, x = p & 31;
        float acc = ab;
        for (int cin = 0; cin < 32; ++cin) {
            float pv = posv[cin];
            const float* vch = vimg + (long)cin * NPIX;
            const float* wc = &w[cin * 9];
#pragma unroll
            for (int tl = 0; tl < 9; ++tl) {
                int gy = y + tl / 3 - 1, gx = x + tl % 3 - 1;
                if (gy >= 0 && gy < HWDIM && gx >= 0 && gx < HWDIM)
                    acc += wc[tl] * (vch[gy * HWDIM + gx] + pv);
            }
        }
        S[((long)(b * NHEADS + hd) * TT + t) * NPIX + p] = acc;
    }
}

// Softmax over t + weighted sum of v -> per-(b,head) output (lq-independent).
__global__ void attn_softmax_apply(const float* __restrict__ S,
                                   const float* __restrict__ V,
                                   float* __restrict__ Oh) {
    int hd = blockIdx.x & 3, b = blockIdx.x >> 2;
    const float* sb = S + (long)(b * NHEADS + hd) * TT * NPIX;
    const float* vb = V + ((long)b * TT * CCH + hd * DHC) * NPIX;
    float* ob = Oh + ((long)b * CCH + hd * DHC) * NPIX;
    for (int p = threadIdx.x; p < NPIX; p += blockDim.x) {
        float s[TT];
        float m = -3.4e38f;
#pragma unroll
        for (int t = 0; t < TT; ++t) { s[t] = sb[(long)t * NPIX + p]; m = fmaxf(m, s[t]); }
        float sum = 0.f;
#pragma unroll
        for (int t = 0; t < TT; ++t) { s[t] = __expf(s[t] - m); sum += s[t]; }
        float inv = 1.f / sum;
        for (int cr = 0; cr < DHC; ++cr) {
            float acc = 0.f;
#pragma unroll
            for (int t = 0; t < TT; ++t)
                acc += s[t] * vb[((long)t * CCH + cr) * NPIX + p];
            ob[(long)cr * NPIX + p] = acc * inv;
        }
    }
}

// A = X + Oh (head output broadcast over lq); also emit f16 copy for ff1.
__global__ void add_residual(const float* __restrict__ X, const float* __restrict__ Oh,
                             float* __restrict__ A, _Float16* __restrict__ Ah, int n) {
    int idx = blockIdx.x * blockDim.x + threadIdx.x;
    if (idx >= n) return;
    int p = idx & (NPIX - 1);
    int c = (idx >> 10) & (CCH - 1);
    int b = (idx >> 17) / TT;
    float v = X[idx] + Oh[((long)b * CCH + c) * NPIX + p];
    A[idx]  = v;
    Ah[idx] = (_Float16)v;
}

// GroupNorm (4 groups of 32 ch): one block per (image, group).
__global__ void group_norm(const float* __restrict__ F2,
                           const float* __restrict__ gw, const float* __restrict__ gb,
                           float* __restrict__ Xout, _Float16* __restrict__ Xh) {
    __shared__ float rs[256], rs2[256];
    const int img = blockIdx.x, g = blockIdx.y;
    const long base = ((long)img * CCH + g * 32) * NPIX;
    const int n = 32 * NPIX;
    float s = 0.f, s2 = 0.f;
    for (int i = threadIdx.x; i < n; i += blockDim.x) {
        float v = F2[base + i];
        s += v; s2 += v * v;
    }
    rs[threadIdx.x] = s; rs2[threadIdx.x] = s2;
    __syncthreads();
    for (int o = 128; o > 0; o >>= 1) {
        if ((int)threadIdx.x < o) { rs[threadIdx.x] += rs[threadIdx.x + o];
                                    rs2[threadIdx.x] += rs2[threadIdx.x + o]; }
        __syncthreads();
    }
    float mu  = rs[0] / (float)n;
    float var = rs2[0] / (float)n - mu * mu;
    float inv = rsqrtf(var + 1e-5f);
    for (int i = threadIdx.x; i < n; i += blockDim.x) {
        int c = g * 32 + (i >> 10);
        float v = (F2[base + i] - mu) * inv * gw[c] + gb[c];
        Xout[base + i] = v;
        Xh[base + i]   = (_Float16)v;
    }
}

// ---------------------------------------------------------------------------
extern "C" void kernel_launch(void* const* d_in, const int* in_sizes, int n_in,
                              void* d_out, int out_size, void* d_ws, size_t ws_size,
                              hipStream_t stream) {
    (void)in_sizes; (void)n_in; (void)out_size; (void)ws_size;

    const float* x_in  = (const float*)d_in[0];
    // d_in[1]=q_w, d_in[2]=q_b: provably dead (cancel in the lk-softmax).
    const float* kv_w  = (const float*)d_in[3];
    const float* kv_b  = (const float*)d_in[4];
    const float* att_w = (const float*)d_in[5];
    const float* att_b = (const float*)d_in[6];
    const float* ff1_w = (const float*)d_in[7];
    const float* ff1_b = (const float*)d_in[8];
    const float* ff2_w = (const float*)d_in[9];
    const float* ff2_b = (const float*)d_in[10];
    const float* gn_w  = (const float*)d_in[11];
    const float* gn_b  = (const float*)d_in[12];
    float* out = (float*)d_out;

    // Workspace bump allocator (256B aligned).
    char* wsp = (char*)d_ws;
    auto alloc = [&](size_t bytes) -> void* {
        void* r = (void*)wsp;
        wsp += (bytes + 255) & ~(size_t)255;
        return r;
    };
    _Float16* Xh   = (_Float16*)alloc((size_t)TOT_ELEMS * 2);
    float*    Xbuf = (float*)   alloc((size_t)TOT_ELEMS * 4);
    float*    V    = (float*)   alloc((size_t)TOT_ELEMS * 4);
    float*    S    = (float*)   alloc((size_t)BB * NHEADS * TT * NPIX * 4);
    float*    Oh   = (float*)   alloc((size_t)BB * CCH * NPIX * 4);
    float*    A    = (float*)   alloc((size_t)TOT_ELEMS * 4);
    _Float16* Ah   = (_Float16*)alloc((size_t)TOT_ELEMS * 2);
    _Float16* F1h  = (_Float16*)alloc((size_t)TOT_ELEMS * 2);
    float*    F2   = (float*)   alloc((size_t)TOT_ELEMS * 4);
    _Float16* wpkv = (_Float16*)alloc((size_t)NLAYERS * NHEADS * 2 * 9 * 512 * 2);
    _Float16* wpf1 = (_Float16*)alloc((size_t)NLAYERS * 8 * 4 * 9 * 512 * 2);
    _Float16* wpf2 = (_Float16*)alloc((size_t)NLAYERS * 8 * 4 * 9 * 512 * 2);

    // Pre-pack weights into WMMA A-fragment order.
    {
        int tH = NLAYERS * NHEADS * 2 * 9 * 512;
        pack_head_w<<<(tH + 255) / 256, 256, 0, stream>>>(kv_w, wpkv);
        int tF = NLAYERS * 8 * 4 * 9 * 512;
        pack_ff_w<<<(tF + 255) / 256, 256, 0, stream>>>(ff1_w, wpf1);
        pack_ff_w<<<(tF + 255) / 256, 256, 0, stream>>>(ff2_w, wpf2);
        f32_to_f16<<<(TOT_ELEMS + 255) / 256, 256, 0, stream>>>(x_in, Xh, TOT_ELEMS);
    }

    for (int li = 0; li < NLAYERS; ++li) {
        const float* Xcur = (li == 0) ? x_in : Xbuf;

        // v = conv(x_head, kv_w) + kv_b, per head (32->32). 2 waves/block.
        for (int hd = 0; hd < NHEADS; ++hd) {
            long ho = (long)hd * DHC * NPIX;
            conv3x3_wmma<<<dim3(NIMG, 1, 64), 64, 0, stream>>>(
                Xh + ho, IMG_STRIDE,
                (const v16h*)(wpkv + (long)(li * NHEADS + hd) * 2 * 9 * 512),
                kv_b + (li * NHEADS + hd) * DHC,
                V + ho, nullptr, nullptr, IMG_STRIDE,
                /*cinBlocks=*/1, /*relu=*/0);
        }

        // Attention logits (k-half conv of v+pos), softmax over t, apply to v.
        attn_logits<<<dim3(BB * NHEADS * TT), 256, 0, stream>>>(V, att_w, att_b, S, li);
        attn_softmax_apply<<<dim3(BB * NHEADS), 256, 0, stream>>>(S, V, Oh);

        // att_out + residual (broadcast over lq), f32 + f16 copies.
        add_residual<<<(TOT_ELEMS + 255) / 256, 256, 0, stream>>>(Xcur, Oh, A, Ah, TOT_ELEMS);

        // ff1: 128->128 conv + bias + relu -> f16. 4 waves/block.
        conv3x3_wmma<<<dim3(NIMG, 2, 64), 128, 0, stream>>>(
            Ah, IMG_STRIDE,
            (const v16h*)(wpf1 + (long)li * 8 * 4 * 9 * 512),
            ff1_b + li * CCH,
            nullptr, F1h, nullptr, IMG_STRIDE,
            /*cinBlocks=*/4, /*relu=*/1);

        // ff2: 128->128 conv + bias + residual(A) -> f32.
        conv3x3_wmma<<<dim3(NIMG, 2, 64), 128, 0, stream>>>(
            F1h, IMG_STRIDE,
            (const v16h*)(wpf2 + (long)li * 8 * 4 * 9 * 512),
            ff2_b + li * CCH,
            F2, nullptr, A, IMG_STRIDE,
            /*cinBlocks=*/4, /*relu=*/0);

        // GroupNorm -> next-layer X (f32 + f16). Last layer writes d_out.
        float* Xnext = (li == NLAYERS - 1) ? out : Xbuf;
        group_norm<<<dim3(NIMG, NHEADS), 256, 0, stream>>>(
            F2, gn_w + li * CCH, gn_b + li * CCH, Xnext, Xh);
    }
}
